// MelBandRoFormer_78769700209082
// MI455X (gfx1250) — matrix operand order repaired and checked by reading, not compiled, not run on hardware
//
#include <hip/hip_runtime.h>
#include <hip/hip_bf16.h>
#include <cmath>

typedef __attribute__((ext_vector_type(16))) _Float16 v16h;
typedef __attribute__((ext_vector_type(8)))  _Float16 h8;
typedef __attribute__((ext_vector_type(8)))  float    v8f;

#define SRATE 44100
#define NFFT  2048
#define NBAND 32
#define DIM   128
#define NLAYER 6
#define NHEAD 4
#define HD    32
#define BATCH 4
#define TT    512
#define NFREQ 1025
#define NTOK  65536   // BATCH*NBAND*TT == BATCH*TT*NBAND

struct BandTable { int s[NBAND]; int e[NBAND]; int maxw; };

// ---------------------------------------------------------------- WMMA helper
__device__ __forceinline__ v8f wmma16(const v16h a, const v16h b, v8f c) {
  // v_wmma_f32_16x16x32_f16  (neg_a, A, neg_b, B, c_mod, C, reuse_a, reuse_b)
  return __builtin_amdgcn_wmma_f32_16x16x32_f16(false, a, false, b, (short)0, c,
                                                false, false);
}
// A-fragment K index for element e (contiguous order: k = 16*hi + e, matches B fills)
__device__ __forceinline__ int kA(int e, bool hi) {
  return e + (hi ? 16 : 0);
}

// ---------------------------------------------------------------- zero fill
__global__ void zero_kernel(float* __restrict__ p, size_t n) {
  size_t i = (size_t)blockIdx.x * 256 + threadIdx.x;
  if (i < n) p[i] = 0.0f;
}

// ---------------------------------------------------------------- band split (VALU)
__global__ __launch_bounds__(128)
void band_in_kernel(const float* __restrict__ re, const float* __restrict__ im,
                    const float* __restrict__ W, const float* __restrict__ bias,
                    float* __restrict__ x, BandTable bt) {
  __shared__ float bv[512];
  int row  = blockIdx.x;            // (b*NBAND + band)*TT + t
  int t    = row & (TT - 1);
  int band = (row >> 9) & (NBAND - 1);
  int b    = row >> 14;
  int s = bt.s[band], w = bt.e[band] - s;
  int tid = threadIdx.x;
  for (int j = tid; j < 2 * w; j += 128) {
    int bin = s + (j >> 1);
    const float* src = (j & 1) ? im : re;
    bv[j] = src[((size_t)b * NFREQ + bin) * TT + t];
  }
  __syncthreads();
  int d = tid;
  float acc = bias[band * DIM + d];
  const float* wp = W + (size_t)band * (2 * bt.maxw) * DIM + d;
  for (int j = 0; j < 2 * w; ++j) acc += bv[j] * wp[(size_t)j * DIM];
  x[(size_t)row * DIM + d] = acc;
}

// ---------------------------------------------------------------- LayerNorm (+RoPE, +cross transpose), f16 out
template <int ROPE, int CROSS>
__global__ __launch_bounds__(256)
void ln_kernel(const float* __restrict__ x, const float* __restrict__ g,
               const float* __restrict__ bb, _Float16* __restrict__ out) {
  int wave = threadIdx.x >> 5, lane = threadIdx.x & 31;
  int token = blockIdx.x * 8 + wave;          // one token per wave, D=128
  const float* xr = x + (size_t)token * DIM;
  float v[4];
#pragma unroll
  for (int i = 0; i < 4; ++i) v[i] = xr[lane * 4 + i];
  float s = v[0] + v[1] + v[2] + v[3];
#pragma unroll
  for (int m = 16; m >= 1; m >>= 1) s += __shfl_xor(s, m, 32);
  float mean = s * (1.0f / 128.0f);
  float q = 0.0f;
#pragma unroll
  for (int i = 0; i < 4; ++i) { float d = v[i] - mean; q += d * d; }
#pragma unroll
  for (int m = 16; m >= 1; m >>= 1) q += __shfl_xor(q, m, 32);
  float rinv = rsqrtf(q * (1.0f / 128.0f) + 1e-5f);
  float y[4];
#pragma unroll
  for (int i = 0; i < 4; ++i) {
    int d = lane * 4 + i;
    y[i] = (v[i] - mean) * rinv * g[d] + bb[d];
  }
  if (ROPE) {
    int t = token & (TT - 1);
    float o[4];
#pragma unroll
    for (int i = 0; i < 4; ++i) {
      int d = lane * 4 + i;
      // inv_freq index = d & 63 ; angle = t * 10000^(-(d&63)/64)
      float ang = (float)t * __expf(-9.2103403719761836f * (float)(d & 63) * (1.0f / 64.0f));
      float rot = (d & 1) ? y[i - 1] : -y[i + 1];   // pairs live inside the 4-block
      o[i] = y[i] * __cosf(ang) + rot * __sinf(ang);
    }
#pragma unroll
    for (int i = 0; i < 4; ++i) y[i] = o[i];
  }
  size_t orow;
  if (CROSS) {  // token = (b*NBAND+nb)*TT + t  ->  (b*TT+t)*NBAND + nb
    int b2 = token >> 14;
    int nb = (token >> 9) & (NBAND - 1);
    int t  = token & (TT - 1);
    orow = (size_t)((b2 * TT + t) * NBAND + nb);
  } else {
    orow = (size_t)token;
  }
  _Float16* op = out + orow * DIM;
#pragma unroll
  for (int i = 0; i < 4; ++i) op[lane * 4 + i] = (_Float16)y[i];
}

// ---------------------------------------------------------------- WMMA GEMM: C = A(f16)[M,K] * W(f32)[K,N] + bias
// MODE 0: f16 out | 1: GELU->f16 out | 2: residual add into f32 | 3: cross scatter-add into f32 (x layout)
template <int MODE>
__global__ __launch_bounds__(128)
void gemm_wmma_kernel(const _Float16* __restrict__ A, const float* __restrict__ W,
                      const float* __restrict__ bias, _Float16* __restrict__ outH,
                      float* __restrict__ outF, int M, int N, int K) {
  __shared__ __align__(16) _Float16 As[64][32];   // row-major A tile
  __shared__ __align__(16) _Float16 Bs[64][32];   // TRANSPOSED weight tile: Bs[n][k]
  const int tid = threadIdx.x;
  const int lane = tid & 31, wave = tid >> 5;
  const int m0 = blockIdx.y * 64, n0 = blockIdx.x * 64;
  const bool hi = lane >= 16;
  const int ln = lane & 15;
  v8f acc[4];
#pragma unroll
  for (int i = 0; i < 4; ++i) acc[i] = {};
  for (int k0 = 0; k0 < K; k0 += 32) {
    __syncthreads();
    // stage A: 64x32 f16, vectorized 16B chunks (global_load_b128 -> ds_store_b128)
#pragma unroll
    for (int ch = 0; ch < 2; ++ch) {
      int idx = tid + ch * 128;            // 256 chunks of 8 halves
      int r = idx >> 2, c0 = (idx & 3) * 8;
      *(h8*)&As[r][c0] = *(const h8*)&A[(size_t)(m0 + r) * K + k0 + c0];
    }
    // stage B transposed: thread -> (n = tid>>1, k-half = (tid&1)*16)
    {
      int n = tid >> 1, kh = (tid & 1) * 16;
      h8 b0, b1;
#pragma unroll
      for (int j = 0; j < 8; ++j) {
        b0[j] = (_Float16)W[(size_t)(k0 + kh + j) * N + n0 + n];
        b1[j] = (_Float16)W[(size_t)(k0 + kh + 8 + j) * N + n0 + n];
      }
      *(h8*)&Bs[n][kh] = b0;
      *(h8*)&Bs[n][kh + 8] = b1;
    }
    if (k0 + 32 < K) {  // prefetch next K tile (global_prefetch_b8)
      __builtin_prefetch(&A[(size_t)(m0 + (tid >> 1)) * K + k0 + 32 + (tid & 1) * 16], 0, 0);
      __builtin_prefetch(&W[(size_t)(k0 + 32 + (tid & 31)) * N + n0 + (tid >> 5) * 16], 0, 0);
    }
    __syncthreads();
    v16h a;
#pragma unroll
    for (int e = 0; e < 16; ++e) a[e] = As[wave * 16 + ln][kA(e, hi)];
#pragma unroll
    for (int nt = 0; nt < 4; ++nt) {
      v16h b;
#pragma unroll
      for (int e = 0; e < 16; ++e) b[e] = Bs[nt * 16 + ln][e + (hi ? 16 : 0)];
      acc[nt] = wmma16(a, b, acc[nt]);
    }
  }
#pragma unroll
  for (int nt = 0; nt < 4; ++nt) {
#pragma unroll
    for (int r = 0; r < 8; ++r) {
      int m = m0 + wave * 16 + r + (hi ? 8 : 0);
      int n = n0 + nt * 16 + ln;
      float v = acc[nt][r] + bias[n];
      if constexpr (MODE == 1) v = 0.5f * v * (1.0f + erff(v * 0.70710678118654752f));
      if constexpr (MODE == 0 || MODE == 1) {
        outH[(size_t)m * N + n] = (_Float16)v;
      } else if constexpr (MODE == 2) {
        size_t o = (size_t)m * N + n;
        outF[o] = outF[o] + v;
      } else {  // MODE 3: m = (b*TT+t)*NBAND+nb -> x row (b*NBAND+nb)*TT+t
        int b2 = m >> 14;
        int t  = (m >> 5) & (TT - 1);
        int nb = m & (NBAND - 1);
        size_t o = (size_t)((b2 * NBAND + nb) * TT + t) * DIM + n;
        outF[o] = outF[o] + v;
      }
    }
  }
}

// ---------------------------------------------------------------- self-attention (per seq,head block; 2 waves)
__global__ __launch_bounds__(64)
void attn_self_kernel(const _Float16* __restrict__ qkv, _Float16* __restrict__ o) {
  __shared__ __align__(16) _Float16 sc[2][16][512];   // per-wave score strip (32 KB)
  const int lane = threadIdx.x & 31, wave = threadIdx.x >> 5;
  const int seq = blockIdx.x >> 2, head = blockIdx.x & 3;
  const bool hi = lane >= 16;
  const int ln = lane & 15;
  const _Float16* base = qkv + (size_t)seq * TT * 384;
  const int qoff = head * HD, koff = 128 + head * HD, voff = 256 + head * HD;
  for (int qi = 0; qi < 16; ++qi) {
    const int qt = wave + 2 * qi;               // 32 query tiles split over 2 waves
    v16h a;
#pragma unroll
    for (int e = 0; e < 16; ++e)
      a[e] = base[(size_t)(qt * 16 + ln) * 384 + qoff + kA(e, hi)];
    for (int kt = 0; kt < 32; ++kt) {           // S = Q * K^T, k-dim = hd = 32
      v16h b;
#pragma unroll
      for (int e = 0; e < 16; ++e)
        b[e] = base[(size_t)(kt * 16 + ln) * 384 + koff + e + (hi ? 16 : 0)];
      v8f s = {};
      s = wmma16(a, b, s);
#pragma unroll
      for (int r = 0; r < 8; ++r)
        sc[wave][r + (hi ? 8 : 0)][kt * 16 + ln] = (_Float16)(s[r] * 0.17677669529663687f);
    }
    // softmax over 512 keys, 16 rows per wave; each lane owns a contiguous 16-chunk
    for (int r = 0; r < 16; ++r) {
      float vals[16];
      float mx = -1e30f;
#pragma unroll
      for (int j = 0; j < 16; ++j) {
        float vv = (float)sc[wave][r][lane * 16 + j];
        vals[j] = vv;
        mx = fmaxf(mx, vv);
      }
#pragma unroll
      for (int m = 16; m >= 1; m >>= 1) mx = fmaxf(mx, __shfl_xor(mx, m, 32));
      float sum = 0.0f;
#pragma unroll
      for (int j = 0; j < 16; ++j) { vals[j] = __expf(vals[j] - mx); sum += vals[j]; }
#pragma unroll
      for (int m = 16; m >= 1; m >>= 1) sum += __shfl_xor(sum, m, 32);
      float inv = 1.0f / sum;
#pragma unroll
      for (int j = 0; j < 16; ++j) sc[wave][r][lane * 16 + j] = (_Float16)(vals[j] * inv);
    }
    // O = P * V   (k-dim 512 -> 16 WMMAs per 16-wide out tile)
#pragma unroll
    for (int ns = 0; ns < 2; ++ns) {
      v8f accO = {};
      for (int kt = 0; kt < 16; ++kt) {
        v16h pa;
#pragma unroll
        for (int e = 0; e < 16; ++e) pa[e] = sc[wave][ln][kt * 32 + kA(e, hi)];
        v16h vb;
#pragma unroll
        for (int e = 0; e < 16; ++e)
          vb[e] = base[(size_t)(kt * 32 + e + (hi ? 16 : 0)) * 384 + voff + ns * 16 + ln];
        accO = wmma16(pa, vb, accO);
      }
#pragma unroll
      for (int r = 0; r < 8; ++r) {
        int m = qt * 16 + r + (hi ? 8 : 0);
        o[((size_t)seq * TT + m) * DIM + head * HD + ns * 16 + ln] = (_Float16)accO[r];
      }
    }
  }
}

// ---------------------------------------------------------------- cross-band attention (seq len 32; 1 wave per head)
__global__ __launch_bounds__(128)
void attn_cross_kernel(const _Float16* __restrict__ qkv, _Float16* __restrict__ co) {
  __shared__ __align__(16) _Float16 sc[4][32][32];
  const int lane = threadIdx.x & 31, head = threadIdx.x >> 5;
  const int sq = blockIdx.x;                    // over BATCH*TT
  const bool hi = lane >= 16;
  const int ln = lane & 15;
  const _Float16* base = qkv + (size_t)sq * NBAND * 384;
  const int qoff = head * HD, koff = 128 + head * HD, voff = 256 + head * HD;
#pragma unroll
  for (int mt = 0; mt < 2; ++mt) {
    v16h a;
#pragma unroll
    for (int e = 0; e < 16; ++e)
      a[e] = base[(size_t)(mt * 16 + ln) * 384 + qoff + kA(e, hi)];
#pragma unroll
    for (int nt = 0; nt < 2; ++nt) {
      v16h b;
#pragma unroll
      for (int e = 0; e < 16; ++e)
        b[e] = base[(size_t)(nt * 16 + ln) * 384 + koff + e + (hi ? 16 : 0)];
      v8f s = {};
      s = wmma16(a, b, s);
#pragma unroll
      for (int r = 0; r < 8; ++r)
        sc[head][mt * 16 + r + (hi ? 8 : 0)][nt * 16 + ln] =
            (_Float16)(s[r] * 0.17677669529663687f);
    }
  }
  for (int r = 0; r < 32; ++r) {                // softmax, lane == column
    float v = (float)sc[head][r][lane];
    float mx = v;
#pragma unroll
    for (int m = 16; m >= 1; m >>= 1) mx = fmaxf(mx, __shfl_xor(mx, m, 32));
    float ev = __expf(v - mx);
    float sum = ev;
#pragma unroll
    for (int m = 16; m >= 1; m >>= 1) sum += __shfl_xor(sum, m, 32);
    sc[head][r][lane] = (_Float16)(ev / sum);
  }
#pragma unroll
  for (int mt = 0; mt < 2; ++mt) {
    v16h pa;
#pragma unroll
    for (int e = 0; e < 16; ++e) pa[e] = sc[head][mt * 16 + ln][kA(e, hi)];
#pragma unroll
    for (int ns = 0; ns < 2; ++ns) {
      v16h vb;
#pragma unroll
      for (int e = 0; e < 16; ++e)
        vb[e] = base[(size_t)(e + (hi ? 16 : 0)) * 384 + voff + ns * 16 + ln];
      v8f acc = {};
      acc = wmma16(pa, vb, acc);
#pragma unroll
      for (int r = 0; r < 8; ++r)
        co[((size_t)sq * NBAND + mt * 16 + r + (hi ? 8 : 0)) * DIM + head * HD +
           ns * 16 + ln] = (_Float16)acc[r];
    }
  }
}

// ---------------------------------------------------------------- band output heads (VALU + sigmoid scatter)
__global__ __launch_bounds__(128)
void band_out_kernel(const float* __restrict__ x, const float* __restrict__ W,
                     const float* __restrict__ bias, float* __restrict__ out,
                     BandTable bt) {
  __shared__ float pv[128];
  int bid  = blockIdx.x;
  int band = bid >> 11;                 // BATCH*TT = 2048
  int rem  = bid & 2047;
  int b    = rem >> 9;
  int t    = rem & (TT - 1);
  int s = bt.s[band], w = bt.e[band] - s;
  int tid = threadIdx.x;
  pv[tid] = x[((size_t)((b * NBAND + band) * TT + t)) * DIM + tid];
  __syncthreads();
  int mw4 = 4 * bt.maxw;
  for (int c = tid; c < 4 * w; c += 128) {
    float acc = bias[band * mw4 + c];
    const float* wp = W + (size_t)band * DIM * mw4 + c;
    for (int d = 0; d < 128; ++d) acc += pv[d] * wp[(size_t)d * mw4];
    float sig = 1.0f / (1.0f + __expf(-acc));
    int r = c & 1, jw = (c >> 1) % w, p = (c >> 1) / w;
    int plane = p * 2 + r;              // [vm_r, vm_i, im_r, im_i]
    out[(((size_t)plane * BATCH + b) * NFREQ + (s + jw)) * TT + t] = sig;
  }
}

// ---------------------------------------------------------------- host side
static void compute_bands(BandTable& bt) {
  double mel_max = 2595.0 * log10(1.0 + (SRATE / 2.0) / 700.0);
  int bins[NBAND + 1];
  for (int i = 0; i <= NBAND; ++i) {
    double mel = mel_max * (double)i / (double)NBAND;
    double fr = 700.0 * (pow(10.0, mel / 2595.0) - 1.0);
    long v = llrint(fr * (double)NFFT / (double)SRATE);  // round-half-even like np.round
    if (v < 0) v = 0;
    if (v > NFREQ - 1) v = NFREQ - 1;
    bins[i] = (int)v;
  }
  int maxw = 0;
  for (int i = 0; i < NBAND; ++i) {
    int s = bins[i], e = bins[i + 1];
    if (e <= s) e = s + 1;
    if (e > NFREQ) e = NFREQ;
    bt.s[i] = s;
    bt.e[i] = e;
    if (e - s > maxw) maxw = e - s;
  }
  bt.maxw = maxw;
}

extern "C" void kernel_launch(void* const* d_in, const int* in_sizes, int n_in,
                              void* d_out, int out_size, void* d_ws, size_t ws_size,
                              hipStream_t stream) {
  const float* stft_re    = (const float*)d_in[0];
  const float* stft_im    = (const float*)d_in[1];
  const float* band_in_W  = (const float*)d_in[2];
  const float* band_in_b  = (const float*)d_in[3];
  const float* band_out_W = (const float*)d_in[4];
  const float* band_out_b = (const float*)d_in[5];
  const float* ln1_g      = (const float*)d_in[6];
  const float* ln1_b      = (const float*)d_in[7];
  const float* attn_in_W  = (const float*)d_in[8];
  const float* attn_in_b  = (const float*)d_in[9];
  const float* attn_out_W = (const float*)d_in[10];
  const float* attn_out_b = (const float*)d_in[11];
  const float* ln2_g      = (const float*)d_in[12];
  const float* ln2_b      = (const float*)d_in[13];
  const float* ff1_W      = (const float*)d_in[14];
  const float* ff1_b      = (const float*)d_in[15];
  const float* ff2_W      = (const float*)d_in[16];
  const float* ff2_b      = (const float*)d_in[17];
  const float* cb_ln_g    = (const float*)d_in[18];
  const float* cb_ln_b    = (const float*)d_in[19];
  const float* cb_in_W    = (const float*)d_in[20];
  const float* cb_in_b    = (const float*)d_in[21];
  const float* cb_out_W   = (const float*)d_in[22];
  const float* cb_out_b   = (const float*)d_in[23];

  BandTable bt;
  compute_bands(bt);

  char* ws = (char*)d_ws;
  size_t off = 0;
  float*    x   = (float*)(ws + off);    off += (size_t)NTOK * DIM * 4;  // 33.5 MB
  _Float16* h   = (_Float16*)(ws + off); off += (size_t)NTOK * DIM * 2;  // 16.8 MB
  _Float16* qkv = (_Float16*)(ws + off); off += (size_t)NTOK * 384 * 2;  // 50.3 MB
  _Float16* o   = (_Float16*)(ws + off); off += (size_t)NTOK * DIM * 2;  // 16.8 MB
  _Float16* u   = (_Float16*)(ws + off); off += (size_t)NTOK * 512 * 2;  // 67.1 MB

  size_t nout = (size_t)out_size;
  zero_kernel<<<(unsigned)((nout + 255) / 256), 256, 0, stream>>>((float*)d_out, nout);

  band_in_kernel<<<NTOK, 128, 0, stream>>>(stft_re, stft_im, band_in_W, band_in_b, x, bt);

  for (int l = 0; l < NLAYER; ++l) {
    ln_kernel<1, 0><<<NTOK / 8, 256, 0, stream>>>(x, ln1_g + l * DIM, ln1_b + l * DIM, h);
    gemm_wmma_kernel<0><<<dim3(6, NTOK / 64), 128, 0, stream>>>(
        h, attn_in_W + (size_t)l * DIM * 384, attn_in_b + (size_t)l * 384,
        qkv, nullptr, NTOK, 384, DIM);
    attn_self_kernel<<<BATCH * NBAND * NHEAD, 64, 0, stream>>>(qkv, o);
    gemm_wmma_kernel<2><<<dim3(2, NTOK / 64), 128, 0, stream>>>(
        o, attn_out_W + (size_t)l * DIM * DIM, attn_out_b + (size_t)l * DIM,
        nullptr, x, NTOK, DIM, DIM);
    ln_kernel<0, 0><<<NTOK / 8, 256, 0, stream>>>(x, ln2_g + l * DIM, ln2_b + l * DIM, h);
    gemm_wmma_kernel<1><<<dim3(8, NTOK / 64), 128, 0, stream>>>(
        h, ff1_W + (size_t)l * DIM * 512, ff1_b + (size_t)l * 512,
        u, nullptr, NTOK, 512, DIM);
    gemm_wmma_kernel<2><<<dim3(2, NTOK / 64), 128, 0, stream>>>(
        u, ff2_W + (size_t)l * 512 * DIM, ff2_b + (size_t)l * DIM,
        nullptr, x, NTOK, DIM, 512);
  }

  // cross-band attention
  ln_kernel<0, 1><<<NTOK / 8, 256, 0, stream>>>(x, cb_ln_g, cb_ln_b, h);
  gemm_wmma_kernel<0><<<dim3(6, NTOK / 64), 128, 0, stream>>>(
      h, cb_in_W, cb_in_b, qkv, nullptr, NTOK, 384, DIM);
  attn_cross_kernel<<<BATCH * TT, 128, 0, stream>>>(qkv, o);
  gemm_wmma_kernel<3><<<dim3(2, NTOK / 64), 128, 0, stream>>>(
      o, cb_out_W, cb_out_b, nullptr, x, NTOK, DIM, DIM);

  band_out_kernel<<<NTOK, 128, 0, stream>>>(x, band_out_W, band_out_b, (float*)d_out, bt);
}
